// CNF_14362370638429
// MI455X (gfx1250) — matrix operand order repaired and checked by reading, not compiled, not run on hardware
//
#include <hip/hip_runtime.h>

typedef __attribute__((ext_vector_type(16))) __bf16       v16bf;
typedef __attribute__((ext_vector_type(8)))  float        v8f;
typedef __attribute__((ext_vector_type(4))) unsigned int u32x4;
typedef __attribute__((ext_vector_type(8)))  int          i32x8;
typedef __attribute__((ext_vector_type(4)))  int          i32x4;

#define Bn   32768
#define Dn   64
#define Hn   256
#define MBLK 128
#define NTH  256

union FragBF { v16bf v; u32x4 q[2]; };

__global__ __launch_bounds__(NTH) void cnf_trace_mlp_kernel(
    const float* __restrict__ x,
    const float* __restrict__ W1, const float* __restrict__ b1,
    const float* __restrict__ W2, const float* __restrict__ b2,
    float* __restrict__ out)
{
    // 32 KB pool: phase 0 = fp32 x-tile landing zone for the TDM;
    // phase 1 = sW1T (8K) + sW2T (8K) + sH (16K). Phases separated by barriers.
    __shared__ __align__(16) unsigned char sPool[MBLK * Dn * 4];   // 32 KB
    __shared__ __align__(16) __bf16 sX[MBLK][Dn];                  // 16 KB
    __shared__ float sS [Hn];                                      // s_k = dot(W1[:,k], W2[k,:])
    __shared__ float sB1[Hn];
    __shared__ float sB2[Dn];
    __shared__ float sTr[MBLK];

    __bf16 (*sW1T)[64] = (__bf16 (*)[64])(sPool);          // [colInChunk][k]
    __bf16 (*sW2T)[64] = (__bf16 (*)[64])(sPool + 8192);   // [outCol][kInChunk]
    __bf16 (*sH)[64]   = (__bf16 (*)[64])(sPool + 16384);  // per-wave h slabs

    const int tid  = threadIdx.x;
    const int l    = tid & 31;       // lane in wave32
    const int w    = tid >> 5;       // wave id (0..7)
    const int half = l >> 4;         // fragment half
    const int ln   = l & 15;
    const long rowBase = (long)blockIdx.x * MBLK;

    // ---------------- phase 0: TDM x-tile load + s_k precompute ----------------
    if (tid < 32) {   // wave 0 issues the tensor DMA (per-wave instruction)
        unsigned long long gaddr =
            (unsigned long long)(uintptr_t)(x + rowBase * (Dn + 1) + 1);
        unsigned ldsOff = (unsigned)(uintptr_t)&sPool[0];   // LDS aperture: addr[31:0]

        u32x4 g0;
        g0[0] = 1u;                                          // count=1, user descriptor
        g0[1] = ldsOff;                                      // lds_addr
        g0[2] = (unsigned)(gaddr & 0xFFFFFFFFull);           // global_addr[31:0]
        g0[3] = (unsigned)((gaddr >> 32) & 0x1FFFFFFull)     // global_addr[56:32]
              | (2u << 30);                                  // type = 2 ("image")

        i32x8 g1;
        g1[0] = (int)(2u << 16);        // data_size = 4 bytes; no mask/pad/iterate
        g1[1] = (int)((unsigned)Dn << 16);      // tensor_dim0 = 64 (bits 79:48 lo)
        g1[2] = (int)((unsigned)Bn << 16);      // tensor_dim1[15:0] = 32768 (bits 111:80 lo)
        g1[3] = (int)((unsigned)Dn << 16);      // tensor_dim1 hi=0 | tile_dim0 = 64
        g1[4] = MBLK;                   // tile_dim1 = 128 rows (tile_dim2 = 0)
        g1[5] = Dn + 1;                 // tensor_dim0_stride = 65 elements
        g1[6] = 0;                      // stride hi / dim1_stride = 0
        g1[7] = 0;

        i32x4 gz4 = {0, 0, 0, 0};            // groups 2/3 unused for 2D tile
        i32x8 gz8 = {0, 0, 0, 0, 0, 0, 0, 0};
        __builtin_amdgcn_tensor_load_to_lds(g0, g1, gz4, gz4, gz8, 0);
    }

    {   // meanwhile (all waves): one s_k per thread (Hn == NTH), biases, trace init
        float s = 0.f;
        #pragma unroll 8
        for (int k = 0; k < Dn; ++k)
            s += W1[k * Hn + tid] * W2[tid * Dn + k];
        sS[tid]  = s;
        sB1[tid] = b1[tid];
    }
    if (tid < Dn)   sB2[tid] = b2[tid];
    if (tid < MBLK) sTr[tid] = 0.f;

    if (tid < 32)
        __builtin_amdgcn_s_wait_tensorcnt(0);   // wave 0: tile landed in LDS
    __syncthreads();

    // convert fp32 tile -> bf16 sX, freeing the pool for phase 1
    {
        const float* sXf = (const float*)sPool;
        for (int idx = tid; idx < MBLK * Dn; idx += NTH)
            sX[idx >> 6][idx & 63] = (__bf16)sXf[idx];
    }
    __syncthreads();

    float tr[8];
    #pragma unroll
    for (int v = 0; v < 8; ++v) tr[v] = 0.f;

    v8f acc2[4];                                  // out accumulators (16x64 per wave)
    #pragma unroll
    for (int n0 = 0; n0 < 4; ++n0) {
        float bias2 = sB2[n0 * 16 + ln];
        #pragma unroll
        for (int v = 0; v < 8; ++v) acc2[n0][v] = bias2;
    }

    // ---------------- phase 1: H in 4 chunks of 64 ----------------
    for (int c = 0; c < Hn / 64; ++c) {
        // stage weight chunks (transposed, bf16)
        for (int idx = tid; idx < 64 * 64; idx += NTH) {
            int k = idx >> 6, col = idx & 63;
            sW1T[col][k] = (__bf16)W1[k * Hn + c * 64 + col];
            sW2T[col][k] = (__bf16)W2[(c * 64 + k) * Dn + col];
        }
        __syncthreads();

        // GEMM1 chunk: h(16x64) = x(16x64) @ W1c(64x64) + b1
        v8f acc1[4];
        #pragma unroll
        for (int n0 = 0; n0 < 4; ++n0) {
            float bias1 = sB1[c * 64 + n0 * 16 + ln];
            #pragma unroll
            for (int v = 0; v < 8; ++v) acc1[n0][v] = bias1;
        }
        #pragma unroll
        for (int k0 = 0; k0 < 64; k0 += 32) {
            FragBF a;   // A 16x32 bf16 layout (ISA 7.12.2)
            const __bf16* arow = &sX[w * 16 + ln][0];
            a.q[0] = *(const u32x4*)(arow + k0 + half * 8);
            a.q[1] = *(const u32x4*)(arow + k0 + 16 + half * 8);
            #pragma unroll
            for (int n0 = 0; n0 < 4; ++n0) {
                FragBF bf;  // B 32x16: same K order as A (k0+half*8.., k0+16+half*8..)
                const __bf16* brow = &sW1T[n0 * 16 + ln][0];
                bf.q[0] = *(const u32x4*)(brow + k0 + half * 8);
                bf.q[1] = *(const u32x4*)(brow + k0 + 16 + half * 8);
                acc1[n0] = __builtin_amdgcn_wmma_f32_16x16x32_bf16(
                    false, a.v, false, bf.v, (short)0, acc1[n0], false, false);
            }
        }

        // tanh + trace partial + h -> LDS (C layout: row = v + half*8, col = n0*16+ln)
        #pragma unroll
        for (int n0 = 0; n0 < 4; ++n0) {
            int colL = n0 * 16 + ln;
            float sk = sS[c * 64 + colL];
            #pragma unroll
            for (int v = 0; v < 8; ++v) {
                float hv = tanhf(acc1[n0][v]);
                tr[v] += (1.f - hv * hv) * sk;
                sH[w * 16 + v + half * 8][colL] = (__bf16)hv;
            }
        }
        // per-wave slab: same-wave DS ops are in-order, no workgroup barrier needed

        // GEMM2 partial: acc2 += h(16x64) @ W2c(64x64)
        #pragma unroll
        for (int k0 = 0; k0 < 64; k0 += 32) {
            FragBF a2;
            const __bf16* arow = &sH[w * 16 + ln][0];
            a2.q[0] = *(const u32x4*)(arow + k0 + half * 8);
            a2.q[1] = *(const u32x4*)(arow + k0 + 16 + half * 8);
            #pragma unroll
            for (int n0 = 0; n0 < 4; ++n0) {
                FragBF bf;
                const __bf16* brow = &sW2T[n0 * 16 + ln][0];
                bf.q[0] = *(const u32x4*)(brow + k0 + half * 8);
                bf.q[1] = *(const u32x4*)(brow + k0 + 16 + half * 8);
                acc2[n0] = __builtin_amdgcn_wmma_f32_16x16x32_bf16(
                    false, a2.v, false, bf.v, (short)0, acc2[n0], false, false);
            }
        }
        __syncthreads();   // protect weight-chunk restage
    }

    // ---------------- trace reduction + output ----------------
    #pragma unroll
    for (int v = 0; v < 8; ++v)
        atomicAdd(&sTr[w * 16 + v + half * 8], tr[v]);   // ds_add_f32, 16 lanes per row

    #pragma unroll
    for (int n0 = 0; n0 < 4; ++n0) {
        #pragma unroll
        for (int v = 0; v < 8; ++v) {
            long rowG = rowBase + w * 16 + v + half * 8;
            out[rowG * (Dn + 1) + 1 + n0 * 16 + ln] = acc2[n0][v];
        }
    }
    __syncthreads();
    if (tid < MBLK)
        out[(rowBase + tid) * (Dn + 1)] = -sTr[tid];
}

extern "C" void kernel_launch(void* const* d_in, const int* in_sizes, int n_in,
                              void* d_out, int out_size, void* d_ws, size_t ws_size,
                              hipStream_t stream) {
    (void)in_sizes; (void)n_in; (void)out_size; (void)d_ws; (void)ws_size;
    const float* x  = (const float*)d_in[0];
    const float* W1 = (const float*)d_in[1];
    const float* b1 = (const float*)d_in[2];
    const float* W2 = (const float*)d_in[3];
    const float* b2 = (const float*)d_in[4];
    float* out = (float*)d_out;

    dim3 grid(Bn / MBLK);   // 256 workgroups
    dim3 block(NTH);        // 8 waves (wave32)
    hipLaunchKernelGGL(cnf_trace_mlp_kernel, grid, block, 0, stream,
                       x, W1, b1, W2, b2, out);
}
